// InfoNCELoss_9680856285466
// MI455X (gfx1250) — matrix operand order repaired and checked by reading, not compiled, not run on hardware
//
#include <hip/hip_runtime.h>
#include <hip/hip_bf16.h>

typedef __attribute__((ext_vector_type(16))) _Float16 v16h;
typedef __attribute__((ext_vector_type(8)))  _Float16 v8h;
typedef __attribute__((ext_vector_type(8)))  float    v8f;
typedef __attribute__((ext_vector_type(4)))  unsigned int u32x4;
typedef __attribute__((ext_vector_type(4)))  int          i32x4;
typedef __attribute__((ext_vector_type(8)))  int          i32x8;

#define B_N   8192
#define D_K   256
#define INV_T (1.0f / 0.07f)
#define KEXP2 (INV_T * 1.442695040888963f)    // log2(e)/T
#define NCG   4                       // column groups (parallelism over N)
#define COLS_PER_CG (B_N / NCG)       // 2048
#define ROWS_PER_BLOCK 128            // 8 waves * 16 rows
#define NUM_ROWBLOCKS (B_N / ROWS_PER_BLOCK)  // 64
#define STAGE_COLS 64
#define NSTAGES (COLS_PER_CG / STAGE_COLS)    // 32
#define LDS_STRIDE 264                // 256 + 8 halfs pad -> conflict-free b128 reads

#if defined(__AMDGCN__) && __has_builtin(__builtin_amdgcn_tensor_load_to_lds) && \
    __has_builtin(__builtin_amdgcn_s_wait_tensorcnt)
#define USE_TDM 1
#else
#define USE_TDM 0
#endif

// Raw v_exp_f32: argument range here is [-41.2, 0] -> no denorm guard needed.
__device__ __forceinline__ float fast_exp2(float x) {
#if defined(__AMDGCN__) && __has_builtin(__builtin_amdgcn_exp2f)
    return __builtin_amdgcn_exp2f(x);
#else
    return exp2f(x);
#endif
}

// Force a wave-uniform branch condition into an SGPR (value is identical
// across the wave by construction; compiler can't prove it).
__device__ __forceinline__ int wave_uniform(int v) {
#if defined(__AMDGCN__) && __has_builtin(__builtin_amdgcn_readfirstlane)
    return __builtin_amdgcn_readfirstlane(v);
#else
    return v;
#endif
}

// ---------------- Kernel 1: L2-normalize rows, write f16 copy ----------------
__global__ __launch_bounds__(256) void k_normalize(const float* __restrict__ x,
                                                   _Float16* __restrict__ fh) {
    const int wave = threadIdx.x >> 5;
    const int lane = threadIdx.x & 31;
    const int row  = blockIdx.x * 8 + wave;
    const float* xr = x + (size_t)row * D_K;

    float4 a = ((const float4*)xr)[lane * 2 + 0];
    float4 b = ((const float4*)xr)[lane * 2 + 1];
    float ss = a.x*a.x + a.y*a.y + a.z*a.z + a.w*a.w
             + b.x*b.x + b.y*b.y + b.z*b.z + b.w*b.w;
#pragma unroll
    for (int m = 1; m <= 16; m <<= 1) ss += __shfl_xor(ss, m, 32);
    const float inv = 1.0f / fmaxf(sqrtf(ss), 1e-12f);

    v8h hv;
    hv[0] = (_Float16)(a.x * inv); hv[1] = (_Float16)(a.y * inv);
    hv[2] = (_Float16)(a.z * inv); hv[3] = (_Float16)(a.w * inv);
    hv[4] = (_Float16)(b.x * inv); hv[5] = (_Float16)(b.y * inv);
    hv[6] = (_Float16)(b.z * inv); hv[7] = (_Float16)(b.w * inv);
    *(v8h*)(fh + (size_t)row * D_K + lane * 8) = hv;
}

#if USE_TDM
// Issue a TDM load of one 64-row x 512B tile (row stride 512B) into LDS,
// with 4 dwords of LDS padding inserted every 128 dwords (-> 264-half stride).
__device__ __forceinline__ void tdm_load_tile(const _Float16* gsrc, unsigned ldsOff) {
    const unsigned long long ga = (unsigned long long)(uintptr_t)gsrc;
    u32x4 g0;
    g0[0] = 1u;                                    // count=1, user descriptor
    g0[1] = ldsOff;                                // lds_addr (bytes)
    g0[2] = (unsigned)(ga & 0xFFFFFFFFu);          // global_addr[31:0]
    g0[3] = (unsigned)((ga >> 32) & 0x01FFFFFFu)   // global_addr[56:32]
          | (2u << 30);                            // type = 2 ("image")
    i32x8 g1;
    g1[0] = (2 << 16)      // data_size = 4B
          | (1 << 20)      // pad_enable
          | (6 << 22)      // pad_interval: 128 dwords
          | (3 << 25);     // pad_amount:   4 dwords
    g1[1] = (128 << 16);   // tensor_dim0[15:0] = 128 dwords (bits 63:48)
    g1[2] = (64  << 16);   // tensor_dim0[31:16]=0 | tensor_dim1[15:0] = 64
    g1[3] = (128 << 16);   // tensor_dim1[31:16]=0 | tile_dim0 = 128 dwords
    g1[4] = 64;            // tile_dim1 = 64 rows | tile_dim2 = 0
    g1[5] = 128;           // tensor_dim0_stride[31:0] = 128 dwords (512B rows)
    g1[6] = 0;             // stride hi | dim1_stride lo (unused for 2D)
    g1[7] = 0;
    const i32x4 gz = {0, 0, 0, 0};
#if __has_include(<hip/amd_detail/amd_gfx1250_TDM.h>)
    const i32x8 gz8 = {0, 0, 0, 0, 0, 0, 0, 0};
    __builtin_amdgcn_tensor_load_to_lds(g0, g1, gz, gz, gz8, 0);   // clang-23 form
#else
    __builtin_amdgcn_tensor_load_to_lds(g0, g1, gz, gz, 0);        // ROCm 7.2 form
#endif
}
#endif

// ------ Kernel 2: WMMA sweep of sim = f f^T with fused online reductions -----
__global__ __launch_bounds__(256) void k_sim(const _Float16* __restrict__ fh,
                                             const int* __restrict__ labels,
                                             float* __restrict__ s1p,
                                             float* __restrict__ s2p,
                                             float* __restrict__ s3p,
                                             float* __restrict__ n2p) {
#if USE_TDM
    __shared__ _Float16 smem[2][STAGE_COLS * LDS_STRIDE];
    __shared__ int      slab[2][STAGE_COLS];
#else
    __shared__ _Float16 smem[1][STAGE_COLS * LDS_STRIDE];
    __shared__ int      slab[1][STAGE_COLS];
#endif

    const int wave = threadIdx.x >> 5;
    const int lane = threadIdx.x & 31;
    const int hi   = lane >> 4;     // 0: lanes 0-15, 1: lanes 16-31
    const int l16  = lane & 15;

    const int blockRow = blockIdx.x & (NUM_ROWBLOCKS - 1);
    const int cg       = blockIdx.x >> 6;
    const int row0     = blockRow * ROWS_PER_BLOCK + wave * 16;
    const int colbase0 = cg * COLS_PER_CG;

    // Preload this wave's 8 A fragments (16 rows x K=256) into VGPRs.
    v16h afrag[8];
    {
        const _Float16* ap = fh + (size_t)(row0 + l16) * D_K;
#pragma unroll
        for (int kk = 0; kk < 8; ++kk) {
            const _Float16* p = ap + kk * 32 + hi * 8;
            union { v16h v; v8h h[2]; } u;
            u.h[0] = *(const v8h*)(p);
            u.h[1] = *(const v8h*)(p + 16);
            afrag[kk] = u.v;
        }
    }
    const int irow_base = row0 + hi * 8;
    int labi[8];
#pragma unroll
    for (int r = 0; r < 8; ++r) labi[r] = labels[irow_base + r];

    // s1: sum of clamped exp; sc2: raw dot sum over positives; n2: positive count;
    // sc3: raw dot sum over all non-self (logit sums recovered in k_finish).
    float s1[8], sc2[8], sc3[8], n2[8];
#pragma unroll
    for (int r = 0; r < 8; ++r) { s1[r] = 0.f; sc2[r] = 0.f; sc3[r] = 0.f; n2[r] = 0.f; }

#if USE_TDM
    // Prologue: DMA stage 0 into buffer 0 (TDM ignores EXEC; issue from wave 0 only).
    if (wave == 0) tdm_load_tile(fh + (size_t)colbase0 * D_K,
                                 (unsigned)(uintptr_t)(void*)&smem[0][0]);
    if (threadIdx.x < STAGE_COLS) slab[0][threadIdx.x] = labels[colbase0 + threadIdx.x];
#endif

    for (int stage = 0; stage < NSTAGES; ++stage) {
        const int colbase = colbase0 + stage * STAGE_COLS;
#if USE_TDM
        const int cur = stage & 1;
        if (wave == 0) __builtin_amdgcn_s_wait_tensorcnt((short)0);  // only issuer waits
        __syncthreads();   // buffer[cur] ready; buffer[1-cur] consumers finished
        if (stage + 1 < NSTAGES) {
            const int nb = colbase + STAGE_COLS;
            if (wave == 0) tdm_load_tile(fh + (size_t)nb * D_K,
                                         (unsigned)(uintptr_t)(void*)&smem[1 - cur][0]);
            if (threadIdx.x < STAGE_COLS) slab[1 - cur][threadIdx.x] = labels[nb + threadIdx.x];
        }
#else
        const int cur = 0;
        __syncthreads();
        {   // Cooperative stage: 64 feature rows -> LDS, 128 B/thread
            const int trow = threadIdx.x >> 2;
            const int tseg = threadIdx.x & 3;
            const _Float16* src = fh + (size_t)(colbase + trow) * D_K + tseg * 64;
            _Float16*       dst = &smem[0][trow * LDS_STRIDE + tseg * 64];
#pragma unroll
            for (int q = 0; q < 8; ++q)
                *(v8h*)(dst + q * 8) = *(const v8h*)(src + q * 8);
            if (threadIdx.x < STAGE_COLS) slab[0][threadIdx.x] = labels[colbase + threadIdx.x];
            const int nbp = (stage + 1 < NSTAGES) ? (colbase + STAGE_COLS) : colbase;
            __builtin_prefetch(fh + (size_t)(nbp + trow) * D_K + tseg * 64, 0, 0);
        }
        __syncthreads();
#endif

#pragma unroll
        for (int nt = 0; nt < 4; ++nt) {
            const int ncol = nt * 16 + l16;
            const _Float16* brow = &smem[cur][ncol * LDS_STRIDE + hi * 8];

            v8f c = {};
#pragma unroll
            for (int kk = 0; kk < 8; ++kk) {
                union { v16h v; v8h h[2]; } ub;
                ub.h[0] = *(const v8h*)(brow + kk * 32);
                ub.h[1] = *(const v8h*)(brow + kk * 32 + 16);
                v16h bfrag = ub.v;
                c = __builtin_amdgcn_wmma_f32_16x16x32_f16(
                        false, afrag[kk], false, bfrag, (short)0, c, false, false);
            }

            const int j    = colbase + ncol;
            const int labj = slab[cur][ncol];
            // Self elements exist only in the diagonal subtile; the test is
            // identical across the wave -> force a scalar (s_cbranch) branch.
            if (wave_uniform(colbase + nt * 16 == row0)) {
#pragma unroll
                for (int r = 0; r < 8; ++r) {
                    const float c0 = c[r];
                    const float ns = (irow_base + r != j) ? 1.0f : 0.0f;
                    const float pm = (labj == labi[r]) ? ns : 0.0f;
                    const float e  = fmaxf(fast_exp2(fmaf(c0, KEXP2, -KEXP2)), 1e-8f);
                    s1[r]  = fmaf(e,  ns, s1[r]);
                    sc3[r] = fmaf(c0, ns, sc3[r]);
                    sc2[r] = fmaf(c0, pm, sc2[r]);
                    n2[r] += pm;
                }
            } else {
#pragma unroll
                for (int r = 0; r < 8; ++r) {
                    const float c0 = c[r];
                    const float pm = (labj == labi[r]) ? 1.0f : 0.0f;
                    const float e  = fmaxf(fast_exp2(fmaf(c0, KEXP2, -KEXP2)), 1e-8f);
                    s1[r] += e;
                    sc3[r] += c0;
                    sc2[r] = fmaf(c0, pm, sc2[r]);
                    n2[r] += pm;
                }
            }
        }
    }

    // Fold across the 16 lanes that share each row (masks 1..8 stay in-half on wave32)
#pragma unroll
    for (int m = 1; m <= 8; m <<= 1) {
#pragma unroll
        for (int r = 0; r < 8; ++r) {
            s1[r]  += __shfl_xor(s1[r],  m, 32);
            sc2[r] += __shfl_xor(sc2[r], m, 32);
            sc3[r] += __shfl_xor(sc3[r], m, 32);
            n2[r]  += __shfl_xor(n2[r],  m, 32);
        }
    }
    if (l16 == 0) {
        const size_t off = (size_t)cg * B_N;
#pragma unroll
        for (int r = 0; r < 8; ++r) {
            const int i = irow_base + r;
            s1p[off + i] = s1[r];
            s2p[off + i] = sc2[r];
            s3p[off + i] = sc3[r];
            n2p[off + i] = n2[r];
        }
    }
}

// ---------------- Kernel 3: chunk fallback + final reduction -----------------
__global__ __launch_bounds__(256) void k_finish(const float* __restrict__ s1p,
                                                const float* __restrict__ s2p,
                                                const float* __restrict__ s3p,
                                                const float* __restrict__ n2p,
                                                float* __restrict__ out) {
    __shared__ float chunk_has[B_N / 256];
    __shared__ float red_l[256];
    __shared__ float red_p[256];
    const int t = threadIdx.x;
    if (t < B_N / 256) chunk_has[t] = 0.0f;
    __syncthreads();

    for (int row = t; row < B_N; row += 256) {
        float n2 = n2p[row] + n2p[B_N + row] + n2p[2 * B_N + row] + n2p[3 * B_N + row];
        if (n2 > 0.0f) chunk_has[row >> 8] = 1.0f;   // same-value race: deterministic
    }
    __syncthreads();

    float tl = 0.0f, tp = 0.0f;
    for (int row = t; row < B_N; row += 256) {
        float s1 = 0.f, sc2 = 0.f, sc3 = 0.f, n2 = 0.f;
#pragma unroll
        for (int g = 0; g < NCG; ++g) {
            s1  += s1p[g * B_N + row];
            sc2 += s2p[g * B_N + row];
            sc3 += s3p[g * B_N + row];
            n2  += n2p[g * B_N + row];
        }
        const float lse = __logf(s1 + 1e-8f);
        // Recover logit sums from raw dot sums: logit = (c - 1) / T
        const float s2 = INV_T * (sc2 - n2);                    // over positives
        const float s3 = INV_T * (sc3 - (float)(B_N - 1));      // over all non-self
        float rl, v;
        if (chunk_has[row >> 8] > 0.0f) {
            if (n2 > 0.0f) { rl = (n2 * lse - s2) / n2; v = 1.0f; }
            else           { rl = 0.0f;                 v = 0.0f; }
        } else {           // chunk had no positives: pos mask falls back to self_mask
            const float np = (float)(B_N - 1);
            rl = (np * lse - s3) / np; v = 1.0f;
        }
        tl += rl; tp += v;
    }
    red_l[t] = tl; red_p[t] = tp;
    __syncthreads();
    for (int s = 128; s > 0; s >>= 1) {
        if (t < s) { red_l[t] += red_l[t + s]; red_p[t] += red_p[t + s]; }
        __syncthreads();
    }
    if (t == 0) {
        const float mean = red_l[0] / (red_p[0] + 1e-8f);
        out[0] = mean;       // loss
        out[1] = red_p[0];   // total_pairs
        out[2] = mean;       // mean_loss
    }
}

// ------------------------------- launcher ------------------------------------
extern "C" void kernel_launch(void* const* d_in, const int* in_sizes, int n_in,
                              void* d_out, int out_size, void* d_ws, size_t ws_size,
                              hipStream_t stream) {
    (void)in_sizes; (void)n_in; (void)out_size; (void)ws_size;
    const float* feat   = (const float*)d_in[0];
    const int*   labels = (const int*)d_in[1];
    float*       out    = (float*)d_out;

    char* ws = (char*)d_ws;
    _Float16* fh = (_Float16*)ws;                               // 4 MB f16 normalized
    float* s1p = (float*)(ws + (size_t)B_N * D_K * 2);          // NCG x B partials
    float* s2p = s1p + (size_t)NCG * B_N;
    float* s3p = s2p + (size_t)NCG * B_N;
    float* n2p = s3p + (size_t)NCG * B_N;

    k_normalize<<<B_N / 8, 256, 0, stream>>>(feat, fh);
    k_sim<<<NUM_ROWBLOCKS * NCG, 256, 0, stream>>>(fh, labels, s1p, s2p, s3p, n2p);
    k_finish<<<1, 256, 0, stream>>>(s1p, s2p, s3p, n2p, out);
}
